// GCN_GAT_85718957294063
// MI455X (gfx1250) — matrix operand order repaired and checked by reading, not compiled, not run on hardware
//
#include <hip/hip_runtime.h>

typedef __attribute__((ext_vector_type(16))) _Float16 v16h;
typedef __attribute__((ext_vector_type(8)))  float    v8f;

// ---------- helpers ----------
static __device__ __forceinline__ unsigned encodeOrd(float f) {
  unsigned u = __float_as_uint(f);
  return (u & 0x80000000u) ? ~u : (u | 0x80000000u);
}
static __device__ __forceinline__ float decodeOrd(unsigned u) {
  return (u & 0x80000000u) ? __uint_as_float(u & 0x7FFFFFFFu)
                           : __uint_as_float(~u);
}
static __device__ __forceinline__ void edge_sd(const int* __restrict__ ei, int E,
                                               int e, int& s, int& d) {
  if (e < E) { s = ei[e]; d = ei[E + e]; }
  else       { s = e - E; d = s; }
}

// ---------- generic fills ----------
__global__ void k_fill_u32(unsigned* __restrict__ p, unsigned v, long n) {
  long i = blockIdx.x * (long)blockDim.x + threadIdx.x;
  if (i < n) p[i] = v;
}

// ---------- degree / norm ----------
__global__ void k_deg_accum(float* __restrict__ deg, const int* __restrict__ ei, int E) {
  int i = blockIdx.x * blockDim.x + threadIdx.x;
  if (i < E) atomicAdd(&deg[ei[E + i]], 1.0f);
}
__global__ void k_rsqrt_inplace(float* __restrict__ p, int n) {
  int i = blockIdx.x * blockDim.x + threadIdx.x;
  if (i < n) p[i] = rsqrtf(p[i]);   // deg >= 1 always (self-loop)
}

// ---------- x @ gcn1_W (K=9, scalar) ----------
__global__ void k_xw(const float* __restrict__ x, const float* __restrict__ W,
                     float* __restrict__ hx, int N) {
  int tid = blockIdx.x * blockDim.x + threadIdx.x;
  if (tid >= N * 32) return;
  int n = tid >> 5, c = tid & 31;
  const float* xr = x + (size_t)n * 9;
  float s = 0.f;
#pragma unroll
  for (int j = 0; j < 9; ++j) s += xr[j] * W[j * 32 + c];
  hx[tid] = s;
}

// ---------- GCN sym-normalized scatter-add (coalesced per-(e,c) mapping) ----------
__global__ void k_gcn_scatter(const float* __restrict__ hin, float* __restrict__ acc,
                              const int* __restrict__ ei, const float* __restrict__ dis,
                              int E, int N, int C) {
  long tid = blockIdx.x * (long)blockDim.x + threadIdx.x;
  long total = (long)(E + N) * C;
  if (tid >= total) return;
  int e = (int)(tid / C), c = (int)(tid % C);
  int s, d; edge_sd(ei, E, e, s, d);
  float norm = dis[s] * dis[d];
  atomicAdd(&acc[(size_t)d * C + c], hin[(size_t)s * C + c] * norm);
}

// ---------- (acc + gcn_b) -> BN -> ReLU -> f16, zero-pad rows >= N ----------
__global__ void k_gcn_bn_relu_f16(const float* __restrict__ acc, const float* __restrict__ b,
                                  const float* __restrict__ gamma, const float* __restrict__ beta,
                                  const float* __restrict__ mean, const float* __restrict__ var,
                                  _Float16* __restrict__ out, int N, int Mpad, int C) {
  long tid = blockIdx.x * (long)blockDim.x + threadIdx.x;
  if (tid >= (long)Mpad * C) return;
  int n = (int)(tid / C), c = (int)(tid % C);
  float v = 0.f;
  if (n < N) {
    v = (acc[tid] + b[c] - mean[c]) * rsqrtf(var[c] + 1e-5f) * gamma[c] + beta[c];
    v = fmaxf(v, 0.f);
  }
  out[tid] = (_Float16)v;
}

// ---------- transpose fp32 W[K x Ncols] -> f16 Bt[Ncols x K] ----------
__global__ void k_transpose_f16(const float* __restrict__ W, _Float16* __restrict__ Bt,
                                int K, int Ncols) {
  long tid = blockIdx.x * (long)blockDim.x + threadIdx.x;
  if (tid >= (long)K * Ncols) return;
  int k = (int)(tid / Ncols), n = (int)(tid % Ncols);
  Bt[(size_t)n * K + k] = (_Float16)W[tid];
}

// ---------- WMMA GEMM: C[Mpad x Ncols] = A[Mpad x K](f16) @ Bt'[Ncols x K](f16) ----------
// ISA layouts (CDNA5 ISA §7.12.2, wave32):
//  A 16x32 f16 : lane l -> M=l&15 ; V0..3 K = (l<16?0:8)+2j ; V4..7 K = 16+(l<16?0:8)+2j
//  B 32x16 f16 : lane l -> N=l&15 ; V0..3 K = (l<16?0:8)+2j ; V4..7 K = 16+(l<16?0:8)+2j
//  C/D 16x16 f32: VGPR v, lane l -> M = v + (l<16?0:8), N = l&15
template <typename OutT>
__global__ void k_wmma_gemm(const _Float16* __restrict__ A, const _Float16* __restrict__ Bt,
                            OutT* __restrict__ C, int Mpad, int K, int Ncols) {
  int wave = (int)((blockIdx.x * (long)blockDim.x + threadIdx.x) >> 5);
  int lane = threadIdx.x & 31;
  int nt = Ncols >> 4;
  int mt = Mpad >> 4;
  if (wave >= mt * nt) return;              // wave-uniform exit
  int tm = wave / nt, tn = wave % nt;
  int r = lane & 15;
  int hi = (lane < 16) ? 0 : 1;
  const _Float16* Arow = A + (size_t)(tm * 16 + r) * K;
  const _Float16* Brow = Bt + (size_t)(tn * 16 + r) * K;
  int akb = hi ? 8 : 0;                     // A half-offset
  int bkb = hi ? 8 : 0;                     // B half-offset
  v8f acc = {};
  for (int k0 = 0; k0 < K; k0 += 32) {
    v16h va, vb;
#pragma unroll
    for (int j = 0; j < 4; ++j) {
      int ka = k0 + akb + 2 * j;            // A VGPR0..3
      va[2 * j]     = Arow[ka];
      va[2 * j + 1] = Arow[ka + 1];
      int kc = k0 + 16 + akb + 2 * j;       // A VGPR4..7
      va[8 + 2 * j]     = Arow[kc];
      va[8 + 2 * j + 1] = Arow[kc + 1];
    }
#pragma unroll
    for (int j = 0; j < 8; ++j) {
      int kb = k0 + (j < 4 ? bkb + 2 * j : 16 + bkb + 2 * (j - 4));  // B VGPR0..7
      vb[2 * j]     = Brow[kb];
      vb[2 * j + 1] = Brow[kb + 1];
    }
    acc = __builtin_amdgcn_wmma_f32_16x16x32_f16(false, va, false, vb,
                                                 (short)0, acc, false, false);
  }
  int rowAdd = hi ? 8 : 0;
#pragma unroll
  for (int v = 0; v < 8; ++v) {
    size_t row = (size_t)(tm * 16 + v + rowAdd);
    C[row * Ncols + tn * 16 + r] = (OutT)acc[v];
  }
}

// ---------- GAT per-node attention terms (CC compile-time -> b128 gathers) ----------
template <typename T, int CC>
__global__ void k_att_terms(const T* __restrict__ hlin, const float* __restrict__ a_src,
                            const float* __restrict__ a_dst, float* __restrict__ asn,
                            float* __restrict__ adn, int N, int H) {
  long tid = blockIdx.x * (long)blockDim.x + threadIdx.x;
  if (tid >= (long)N * H) return;
  int n = (int)(tid / H), h = (int)(tid % H);
  const T* row = hlin + (size_t)n * H * CC + (size_t)h * CC;
  const float* as = a_src + (size_t)h * CC;
  const float* ad = a_dst + (size_t)h * CC;
  float s1 = 0.f, s2 = 0.f;
#pragma unroll
  for (int c = 0; c < CC; ++c) {
    float v = (float)row[c];
    s1 += v * as[c];
    s2 += v * ad[c];
  }
  asn[tid] = s1;
  adn[tid] = s2;
}

// ---------- GAT softmax passes ----------
__global__ void k_gat_pass1(const int* __restrict__ ei, int E, int N, int H,
                            const float* __restrict__ asn, const float* __restrict__ adn,
                            unsigned* __restrict__ mbuf) {
  long tid = blockIdx.x * (long)blockDim.x + threadIdx.x;
  if (tid >= (long)(E + N) * H) return;
  int e = (int)(tid / H), h = (int)(tid % H);
  int s, d; edge_sd(ei, E, e, s, d);
  float ev = asn[(size_t)s * H + h] + adn[(size_t)d * H + h];
  ev = (ev >= 0.f) ? ev : 0.2f * ev;
  atomicMax(&mbuf[(size_t)d * H + h], encodeOrd(ev));
}
__global__ void k_gat_pass2(const int* __restrict__ ei, int E, int N, int H,
                            const float* __restrict__ asn, const float* __restrict__ adn,
                            const unsigned* __restrict__ mbuf, float* __restrict__ sbuf) {
  long tid = blockIdx.x * (long)blockDim.x + threadIdx.x;
  if (tid >= (long)(E + N) * H) return;
  int e = (int)(tid / H), h = (int)(tid % H);
  int s, d; edge_sd(ei, E, e, s, d);
  float ev = asn[(size_t)s * H + h] + adn[(size_t)d * H + h];
  ev = (ev >= 0.f) ? ev : 0.2f * ev;
  float m = decodeOrd(mbuf[(size_t)d * H + h]);
  atomicAdd(&sbuf[(size_t)d * H + h], expf(ev - m));
}
// CC compile-time: 4x b128 gather per lane + clause-grouped atomic adds
template <typename T, int CC>
__global__ void k_gat_pass3(const int* __restrict__ ei, int E, int N, int H,
                            const float* __restrict__ asn, const float* __restrict__ adn,
                            const unsigned* __restrict__ mbuf, const float* __restrict__ sbuf,
                            const T* __restrict__ hlin, float* __restrict__ acc) {
  long tid = blockIdx.x * (long)blockDim.x + threadIdx.x;
  if (tid >= (long)(E + N) * H) return;
  int e = (int)(tid / H), h = (int)(tid % H);
  int s, d; edge_sd(ei, E, e, s, d);
  size_t dh = (size_t)d * H + h;
  float ev = asn[(size_t)s * H + h] + adn[dh];
  ev = (ev >= 0.f) ? ev : 0.2f * ev;
  float alpha = expf(ev - decodeOrd(mbuf[dh])) / (sbuf[dh] + 1e-16f);
  const T* row = hlin + (size_t)s * H * CC + (size_t)h * CC;
  float* out = acc + (size_t)d * H * CC + (size_t)h * CC;
  float vals[CC];
#pragma unroll
  for (int c = 0; c < CC; ++c) vals[c] = (float)row[c] * alpha;
#pragma unroll
  for (int c = 0; c < CC; ++c) atomicAdd(&out[c], vals[c]);
}

// ---------- bias + relu variants ----------
__global__ void k_bias_relu_f16(const float* __restrict__ acc, const float* __restrict__ bias,
                                _Float16* __restrict__ out, int N, int Mpad, int C) {
  long tid = blockIdx.x * (long)blockDim.x + threadIdx.x;
  if (tid >= (long)Mpad * C) return;
  int n = (int)(tid / C), c = (int)(tid % C);
  float v = 0.f;
  if (n < N) v = fmaxf(acc[tid] + bias[c], 0.f);
  out[tid] = (_Float16)v;
}
__global__ void k_bias_relu_f32(const float* __restrict__ acc, const float* __restrict__ bias,
                                float* __restrict__ out, int N, int C) {
  long tid = blockIdx.x * (long)blockDim.x + threadIdx.x;
  if (tid >= (long)N * C) return;
  int c = (int)(tid % C);
  out[tid] = fmaxf(acc[tid] + bias[c], 0.f);
}

// ---------- global max pool + final linear/sigmoid ----------
__global__ void k_pool_max(const float* __restrict__ hfin, const int* __restrict__ batch,
                           unsigned* __restrict__ poolU, int N, int C) {
  long tid = blockIdx.x * (long)blockDim.x + threadIdx.x;
  if (tid >= (long)N * C) return;
  int n = (int)(tid / C), c = (int)(tid % C);
  atomicMax(&poolU[(size_t)batch[n] * C + c], encodeOrd(hfin[tid]));
}
__global__ void k_final(const unsigned* __restrict__ poolU, const float* __restrict__ linW,
                        const float* __restrict__ linb, float* __restrict__ out, int G, int C) {
  int g = blockIdx.x * blockDim.x + threadIdx.x;
  if (g >= G) return;
  float z = linb[0];
  for (int c = 0; c < C; ++c) {
    float p = decodeOrd(poolU[(size_t)g * C + c]);
    if (!(p > -1e38f && p < 1e38f)) p = 0.f;   // non-finite guard
    z += p * linW[c];
  }
  out[g] = 1.f / (1.f + expf(-z));
}

// ---------- host ----------
extern "C" void kernel_launch(void* const* d_in, const int* in_sizes, int n_in,
                              void* d_out, int out_size, void* d_ws, size_t ws_size,
                              hipStream_t stream) {
  const float* x       = (const float*)d_in[0];
  const int*   ei      = (const int*)d_in[1];
  const int*   batch   = (const int*)d_in[2];
  const float* gcn1_W  = (const float*)d_in[3];
  const float* gcn1_b  = (const float*)d_in[4];
  const float* bn1_g   = (const float*)d_in[5];
  const float* bn1_bt  = (const float*)d_in[6];
  const float* bn1_m   = (const float*)d_in[7];
  const float* bn1_v   = (const float*)d_in[8];
  const float* gat1_W  = (const float*)d_in[9];
  const float* gat1_as = (const float*)d_in[10];
  const float* gat1_ad = (const float*)d_in[11];
  const float* gat1_b  = (const float*)d_in[12];
  const float* gcn2_W  = (const float*)d_in[13];
  const float* gcn2_b  = (const float*)d_in[14];
  const float* bn2_g   = (const float*)d_in[15];
  const float* bn2_bt  = (const float*)d_in[16];
  const float* bn2_m   = (const float*)d_in[17];
  const float* bn2_v   = (const float*)d_in[18];
  const float* gat2_W  = (const float*)d_in[19];
  const float* gat2_as = (const float*)d_in[20];
  const float* gat2_ad = (const float*)d_in[21];
  const float* gat2_b  = (const float*)d_in[22];
  const float* lin_W   = (const float*)d_in[23];
  const float* lin_b   = (const float*)d_in[24];

  const int N  = in_sizes[0] / 9;
  const int E  = in_sizes[1] / 2;
  const int C  = 32;
  const int H1 = in_sizes[10] / C;      // 20
  const int F1 = H1 * C;                // 640
  const int G  = out_size;              // 128
  const int Mpad = ((N + 15) / 16) * 16;

  size_t off = 0;
  auto alloc = [&](size_t bytes) -> char* {
    size_t o = (off + 255) & ~(size_t)255;
    off = o + bytes;
    return (char*)d_ws + o;
  };
  float*     dis     = (float*)alloc((size_t)N * 4);
  float*     hx      = (float*)alloc((size_t)N * C * 4);
  float*     acc1    = (float*)alloc((size_t)N * C * 4);
  _Float16*  h1h     = (_Float16*)alloc((size_t)Mpad * C * 2);
  _Float16*  B1t     = (_Float16*)alloc((size_t)F1 * C * 2);
  _Float16*  hlin    = (_Float16*)alloc((size_t)Mpad * F1 * 2);   // reused post-GAT1
  float*     asn     = (float*)alloc((size_t)N * H1 * 4);
  float*     adn     = (float*)alloc((size_t)N * H1 * 4);
  unsigned*  mbuf    = (unsigned*)alloc((size_t)N * H1 * 4);
  float*     sbuf    = (float*)alloc((size_t)N * H1 * 4);
  float*     gatacc  = (float*)alloc((size_t)N * F1 * 4);
  _Float16*  B2t     = (_Float16*)alloc((size_t)C * F1 * 2);
  float*     t2      = (float*)alloc((size_t)Mpad * C * 4);
  float*     acc2    = (float*)alloc((size_t)N * C * 4);
  _Float16*  h2h     = (_Float16*)alloc((size_t)Mpad * C * 2);
  _Float16*  B3t     = (_Float16*)alloc((size_t)C * C * 2);
  float*     hlin2   = (float*)alloc((size_t)Mpad * C * 4);
  float*     as2     = (float*)alloc((size_t)N * 4);
  float*     ad2     = (float*)alloc((size_t)N * 4);
  unsigned*  m2      = (unsigned*)alloc((size_t)N * 4);
  float*     s2      = (float*)alloc((size_t)N * 4);
  float*     g2acc   = (float*)alloc((size_t)N * C * 4);
  float*     hfin    = (float*)alloc((size_t)N * C * 4);
  unsigned*  poolU   = (unsigned*)alloc((size_t)G * C * 4);

  auto gf = [](long n, int bs) { return (unsigned)((n + bs - 1) / bs); };
  const int BS = 256;

  // degree -> dis
  k_fill_u32<<<gf(N, BS), BS, 0, stream>>>((unsigned*)dis, 0x3F800000u, N); // 1.0f
  k_deg_accum<<<gf(E, BS), BS, 0, stream>>>(dis, ei, E);
  k_rsqrt_inplace<<<gf(N, BS), BS, 0, stream>>>(dis, N);

  // GCN1
  k_xw<<<gf((long)N * C, BS), BS, 0, stream>>>(x, gcn1_W, hx, N);
  k_fill_u32<<<gf((long)N * C, BS), BS, 0, stream>>>((unsigned*)acc1, 0u, (long)N * C);
  k_gcn_scatter<<<gf((long)(E + N) * C, BS), BS, 0, stream>>>(hx, acc1, ei, dis, E, N, C);
  k_gcn_bn_relu_f16<<<gf((long)Mpad * C, BS), BS, 0, stream>>>(acc1, gcn1_b, bn1_g, bn1_bt,
                                                               bn1_m, bn1_v, h1h, N, Mpad, C);
  // GAT1 linear (WMMA): [Mpad x 32] @ [32 x 640] -> f16
  k_transpose_f16<<<gf((long)C * F1, BS), BS, 0, stream>>>(gat1_W, B1t, C, F1);
  {
    long waves = (long)(Mpad / 16) * (F1 / 16);
    k_wmma_gemm<_Float16><<<gf(waves * 32, BS), BS, 0, stream>>>(h1h, B1t, hlin, Mpad, C, F1);
  }
  k_att_terms<_Float16, 32><<<gf((long)N * H1, BS), BS, 0, stream>>>(hlin, gat1_as, gat1_ad,
                                                                     asn, adn, N, H1);
  k_fill_u32<<<gf((long)N * H1, BS), BS, 0, stream>>>(mbuf, 0u, (long)N * H1);
  k_gat_pass1<<<gf((long)(E + N) * H1, BS), BS, 0, stream>>>(ei, E, N, H1, asn, adn, mbuf);
  k_fill_u32<<<gf((long)N * H1, BS), BS, 0, stream>>>((unsigned*)sbuf, 0u, (long)N * H1);
  k_gat_pass2<<<gf((long)(E + N) * H1, BS), BS, 0, stream>>>(ei, E, N, H1, asn, adn, mbuf, sbuf);
  k_fill_u32<<<gf((long)N * F1, BS), BS, 0, stream>>>((unsigned*)gatacc, 0u, (long)N * F1);
  k_gat_pass3<_Float16, 32><<<gf((long)(E + N) * H1, BS), BS, 0, stream>>>(ei, E, N, H1, asn, adn,
                                                                           mbuf, sbuf, hlin, gatacc);
  k_bias_relu_f16<<<gf((long)Mpad * F1, BS), BS, 0, stream>>>(gatacc, gat1_b, hlin, N, Mpad, F1);

  // GCN2 linear (WMMA): [Mpad x 640] @ [640 x 32] -> f32
  k_transpose_f16<<<gf((long)F1 * C, BS), BS, 0, stream>>>(gcn2_W, B2t, F1, C);
  {
    long waves = (long)(Mpad / 16) * (C / 16);
    k_wmma_gemm<float><<<gf(waves * 32, BS), BS, 0, stream>>>(hlin, B2t, t2, Mpad, F1, C);
  }
  k_fill_u32<<<gf((long)N * C, BS), BS, 0, stream>>>((unsigned*)acc2, 0u, (long)N * C);
  k_gcn_scatter<<<gf((long)(E + N) * C, BS), BS, 0, stream>>>(t2, acc2, ei, dis, E, N, C);
  k_gcn_bn_relu_f16<<<gf((long)Mpad * C, BS), BS, 0, stream>>>(acc2, gcn2_b, bn2_g, bn2_bt,
                                                               bn2_m, bn2_v, h2h, N, Mpad, C);
  // GAT2 linear (WMMA): [Mpad x 32] @ [32 x 32] -> f32
  k_transpose_f16<<<gf((long)C * C, BS), BS, 0, stream>>>(gat2_W, B3t, C, C);
  {
    long waves = (long)(Mpad / 16) * (C / 16);
    k_wmma_gemm<float><<<gf(waves * 32, BS), BS, 0, stream>>>(h2h, B3t, hlin2, Mpad, C, C);
  }
  k_att_terms<float, 32><<<gf((long)N, BS), BS, 0, stream>>>(hlin2, gat2_as, gat2_ad, as2, ad2, N, 1);
  k_fill_u32<<<gf(N, BS), BS, 0, stream>>>(m2, 0u, N);
  k_gat_pass1<<<gf((long)(E + N), BS), BS, 0, stream>>>(ei, E, N, 1, as2, ad2, m2);
  k_fill_u32<<<gf(N, BS), BS, 0, stream>>>((unsigned*)s2, 0u, N);
  k_gat_pass2<<<gf((long)(E + N), BS), BS, 0, stream>>>(ei, E, N, 1, as2, ad2, m2, s2);
  k_fill_u32<<<gf((long)N * C, BS), BS, 0, stream>>>((unsigned*)g2acc, 0u, (long)N * C);
  k_gat_pass3<float, 32><<<gf((long)(E + N), BS), BS, 0, stream>>>(ei, E, N, 1, as2, ad2, m2, s2,
                                                                   hlin2, g2acc);
  k_bias_relu_f32<<<gf((long)N * C, BS), BS, 0, stream>>>(g2acc, gat2_b, hfin, N, C);

  // pool + head
  k_fill_u32<<<gf((long)G * C, BS), BS, 0, stream>>>(poolU, 0u, (long)G * C);
  k_pool_max<<<gf((long)N * C, BS), BS, 0, stream>>>(hfin, batch, poolU, N, C);
  k_final<<<gf(G, 128), 128, 0, stream>>>(poolU, lin_W, lin_b, (float*)d_out, G, C);
  (void)ws_size; (void)n_in;
}
